// GATHead_21328807592248
// MI455X (gfx1250) — matrix operand order repaired and checked by reading, not compiled, not run on hardware
//
#include <hip/hip_runtime.h>
#include <hip/hip_bf16.h>
#include <stdint.h>

typedef __attribute__((ext_vector_type(16))) _Float16 v16h;
typedef __attribute__((ext_vector_type(8)))  float    v8f;

#define B_   8
#define N_   2048
#define CIN  128
#define COUT 64
#define NEG  0.2f
#define LOG2E 1.44269504088896f

// ---------------------------------------------------------------------------
// Kernel 1: pack A (int 0/1, N x N) into per-row 32-bit masks.
// ---------------------------------------------------------------------------
__global__ __launch_bounds__(256) void k_amask(const int* __restrict__ A,
                                               uint32_t* __restrict__ Amask) {
    int word = blockIdx.x * 256 + threadIdx.x;      // 0 .. 2048*64-1
    const int* arow = A + (size_t)(word >> 6) * N_ + (word & 63) * 32;
    uint32_t m = 0;
#pragma unroll
    for (int t = 0; t < 32; ++t)
        m |= (arow[t] > 0 ? 1u : 0u) << t;
    Amask[word] = m;
}

// ---------------------------------------------------------------------------
// Kernel 2: Wh = x @ W (f32 accumulate), f1 = Wh@a1, f2 = Wh@a2.
//  - Wh stored f16, pre-swizzled into the WMMA B-operand layout.
//  - f2 stored pre-swizzled in the WMMA A-operand K-order AND pre-scaled by
//    log2(e) so the hot loop works directly in the exp2 domain:
//      k(h,g) = h + (h&8) + g*8 ;  f2swL[(b*64+j/32)*32 + g*16+h] = L*f2[j]
// ---------------------------------------------------------------------------
__global__ __launch_bounds__(64) void k_wh(const float* __restrict__ x,
                                           const float* __restrict__ W,
                                           const float* __restrict__ attn,
                                           float* __restrict__ f1,
                                           float* __restrict__ f2,
                                           float* __restrict__ f2swL,
                                           _Float16* __restrict__ packed) {
    __shared__ float sW[CIN * COUT];
    for (int idx = threadIdx.x; idx < CIN * COUT; idx += 64)
        sW[idx] = W[idx];
    __syncthreads();

    int row = blockIdx.x * 64 + threadIdx.x;        // 0 .. B*N-1
    int b = row >> 11;
    int n = row & (N_ - 1);
    const float* xr = x + (size_t)row * CIN;

    float wh[COUT];
#pragma unroll
    for (int o = 0; o < COUT; ++o) wh[o] = 0.0f;

    for (int k = 0; k < CIN; ++k) {
        float xv = xr[k];
        const float* wr = sW + k * COUT;
#pragma unroll
        for (int o = 0; o < COUT; ++o)
            wh[o] = fmaf(xv, wr[o], wh[o]);
    }

    float s1 = 0.0f, s2 = 0.0f;
#pragma unroll
    for (int o = 0; o < COUT; ++o) {
        s1 = fmaf(wh[o], attn[o], s1);
        s2 = fmaf(wh[o], attn[COUT + o], s2);
    }
    f1[row] = s1;
    f2[row] = s2;

    int jt32 = n >> 5;
    int w    = n & 31;
    int g    = (w >> 3) & 1;    // lane-group for B layout (K = h + (h&8) + g*8)
    int h    = (w & 7) + 8 * (w >> 4);
    // A-layout swizzle of f2 (same g/h mapping as B)
    int ga   = (w >> 3) & 1;
    int ha   = w - 8 * ((w >> 3) & 1) - 8 * (w >> 4);
    f2swL[(size_t)(b * 64 + jt32) * 32 + ga * 16 + ha] = s2 * LOG2E;

    size_t tbase = (size_t)(b * 64 + jt32) * 4;
#pragma unroll
    for (int oc = 0; oc < 4; ++oc)
#pragma unroll
        for (int nc = 0; nc < 16; ++nc)
            packed[((tbase + oc) * 32 + (g * 16 + nc)) * 16 + h] =
                (_Float16)wh[oc * 16 + nc];
}

// ---------------------------------------------------------------------------
// Kernel 3: per-batch max of f2 (softmax shift bound).
// ---------------------------------------------------------------------------
__global__ __launch_bounds__(256) void k_f2max(const float* __restrict__ f2,
                                               float* __restrict__ f2max) {
    __shared__ float red[8];
    int b = blockIdx.x;
    int t = threadIdx.x;
    const float* p = f2 + (size_t)b * N_;
    float m = -3.0e38f;
    for (int j = t; j < N_; j += 256)
        m = fmaxf(m, p[j]);
#pragma unroll
    for (int off = 16; off > 0; off >>= 1)
        m = fmaxf(m, __shfl_xor(m, off, 32));
    if ((t & 31) == 0) red[t >> 5] = m;
    __syncthreads();
    if (t == 0) {
        float mm = red[0];
#pragma unroll
        for (int i = 1; i < 8; ++i) mm = fmaxf(mm, red[i]);
        f2max[b] = mm;
    }
}

// ---------------------------------------------------------------------------
// Kernel 4: SINGLE-PASS fused masked-softmax aggregation.
// One wave per (b, 16-row tile).  m_r = leaky(f1_r + max_j f2_j) bounds every
// masked logit (leaky monotone); softmax is shift-invariant so no row max pass.
// Row normalizer computed by a 5th WMMA against an all-ones B tile: its
// accumulator lands in exactly the C-layout register the epilogue needs.
// exp2-domain identity:  L*(leaky(x)-m) = max(L*f2 + c1, fma(0.2, L*f2, c2))
//   with c1 = L*(f1-m), c2 = L*(0.2*f1-m).
// ---------------------------------------------------------------------------
__global__ __launch_bounds__(32) void k_gat(const float* __restrict__ f1,
                                            const float* __restrict__ f2swL,
                                            const float* __restrict__ f2max,
                                            const _Float16* __restrict__ packed,
                                            const uint32_t* __restrict__ Amask,
                                            float* __restrict__ out) {
    __shared__ uint32_t sMask[16 * 64];             // 4 KB mask words, rows i0..i0+15

    int bid  = blockIdx.x;            // 0 .. 1023
    int b    = bid >> 7;
    int i0   = (bid & 127) * 16;
    int lane = threadIdx.x;           // wave32
    int r    = lane & 15;
    int g    = lane >> 4;

    for (int idx = lane; idx < 16 * 64; idx += 32)
        sMask[idx] = Amask[(size_t)(i0 + (idx >> 6)) * 64 + (idx & 63)];
    __syncthreads();

    float f1v = f1[(size_t)b * N_ + i0 + r];
    float mv  = f1v + f2max[b];
    mv = fmaxf(mv, NEG * mv);                        // upper bound on row logits
    float c1 = (f1v - mv) * LOG2E;
    float c2 = (NEG * f1v - mv) * LOG2E;

    v8f acc0 = {}, acc1 = {}, acc2 = {}, acc3 = {}, acc4 = {};
    v16h ones;
#pragma unroll
    for (int h = 0; h < 16; ++h) ones[h] = (_Float16)1.0f;

    const float4* fswb = (const float4*)(f2swL + (size_t)b * 64 * 32);

    for (int jt32 = 0; jt32 < 64; ++jt32) {
        // B tiles first: two b128 loads each; latency hidden by p-tile math.
        const _Float16* pk = packed + (size_t)(b * 64 + jt32) * 4 * 32 * 16;
        v16h b0 = *(const v16h*)(pk + ((size_t)(0 * 32 + lane)) * 16);
        v16h b1 = *(const v16h*)(pk + ((size_t)(1 * 32 + lane)) * 16);
        v16h b2 = *(const v16h*)(pk + ((size_t)(2 * 32 + lane)) * 16);
        v16h b3 = *(const v16h*)(pk + ((size_t)(3 * 32 + lane)) * 16);

        // 16 pre-scaled f2 values in A-layout order for this half-wave.
        const float4* fsw = fswb + (size_t)jt32 * 8 + g * 4;
        float fv[16];
#pragma unroll
        for (int q = 0; q < 4; ++q) {
            float4 t = fsw[q];
            fv[q * 4 + 0] = t.x; fv[q * 4 + 1] = t.y;
            fv[q * 4 + 2] = t.z; fv[q * 4 + 3] = t.w;
        }

        // 16-bit mask, bit h <-> k(h,g) = h + (h&8) + g*8
        uint32_t aw  = sMask[r * 64 + jt32];
        uint32_t msk = ((aw >> (g * 8)) & 0xFFu) |
                       (((aw >> (16 + g * 8)) & 0xFFu) << 8);

        v16h a;
#pragma unroll
        for (int h = 0; h < 16; ++h) {
            float arg = fmaxf(fv[h] + c1, fmaf(NEG, fv[h], c2)); // add+fma+max
            float pe  = __builtin_amdgcn_exp2f(arg);             // v_exp_f32
            a[h] = (_Float16)(((msk >> h) & 1u) ? pe : 0.0f);    // v_cndmask
        }

        acc0 = __builtin_amdgcn_wmma_f32_16x16x32_f16(false, a, false, b0,
                                                      (short)0, acc0, false, false);
        acc1 = __builtin_amdgcn_wmma_f32_16x16x32_f16(false, a, false, b1,
                                                      (short)0, acc1, false, false);
        acc2 = __builtin_amdgcn_wmma_f32_16x16x32_f16(false, a, false, b2,
                                                      (short)0, acc2, false, false);
        acc3 = __builtin_amdgcn_wmma_f32_16x16x32_f16(false, a, false, b3,
                                                      (short)0, acc3, false, false);
        acc4 = __builtin_amdgcn_wmma_f32_16x16x32_f16(false, a, false, ones,
                                                      (short)0, acc4, false, false);
    }

    // ---------------- Epilogue: acc4[v] == row normalizer l_{g*8+v} ---------
    float* outg = out + ((size_t)b * N_ + i0) * COUT;
    int col = lane & 15;
#pragma unroll
    for (int v = 0; v < 8; ++v) {
        float linv = __builtin_amdgcn_rcpf(fmaxf(acc4[v], 1e-37f));
        int M = g * 8 + v;                        // C/D layout: VGPR v -> row g*8+v
        float* orow = outg + (size_t)M * COUT;
        orow[ 0 + col] = acc0[v] * linv;
        orow[16 + col] = acc1[v] * linv;
        orow[32 + col] = acc2[v] * linv;
        orow[48 + col] = acc3[v] * linv;
    }
}

// ---------------------------------------------------------------------------
extern "C" void kernel_launch(void* const* d_in, const int* in_sizes, int n_in,
                              void* d_out, int out_size, void* d_ws, size_t ws_size,
                              hipStream_t stream) {
    const float* x    = (const float*)d_in[0];   // [8, 2048, 128]
    const int*   A    = (const int*)d_in[1];     // [2048, 2048]
    const float* W    = (const float*)d_in[2];   // [128, 64]
    const float* attn = (const float*)d_in[3];   // [128, 1]
    float*       out  = (float*)d_out;           // [8, 2048, 64]

    uint8_t* ws = (uint8_t*)d_ws;
    float*     f1     = (float*)(ws);                          //  64 KB
    float*     f2     = (float*)(ws + 65536);                  //  64 KB
    float*     f2swL  = (float*)(ws + 131072);                 //  64 KB
    float*     f2max  = (float*)(ws + 196608);                 // 256 B (8 used)
    _Float16*  packed = (_Float16*)(ws + 196864);              //   2 MB
    uint32_t*  Amask  = (uint32_t*)(ws + 196864 + 2097152);    // 512 KB

    k_amask<<<dim3((N_ * 64) / 256), dim3(256), 0, stream>>>(A, Amask);
    k_wh   <<<dim3((B_ * N_) / 64),  dim3(64),  0, stream>>>(x, W, attn, f1, f2, f2swL, packed);
    k_f2max<<<dim3(B_),              dim3(256), 0, stream>>>(f2, f2max);
    k_gat  <<<dim3(B_ * (N_ / 16)),  dim3(32),  0, stream>>>(f1, f2swL, f2max, packed, Amask, out);
}
